// CrossAttention_68487548502663
// MI455X (gfx1250) — matrix operand order repaired and checked by reading, not compiled, not run on hardware
//
#include <hip/hip_runtime.h>
#include <hip/hip_bf16.h>

// CDNA5 / gfx1250 wave32 WMMA implementation of cross-attention.
// Fixed problem sizes from the reference: s1=s2=2048, d1=d2=1024, H=16, DK=DV=64.

typedef __attribute__((ext_vector_type(16))) _Float16 v16h;
typedef __attribute__((ext_vector_type(8)))  _Float16 v8h;
typedef __attribute__((ext_vector_type(4)))  _Float16 v4h;
typedef __attribute__((ext_vector_type(8)))  float    v8f;

#define S1 2048
#define S2 2048
#define DMODEL 1024
#define NHEAD 16
#define DHEAD 64      // DK == DV == 64
#define HD 1024       // NHEAD * DHEAD

// ---------------------------------------------------------------------------
// WMMA helpers (layouts per CDNA5 ISA, wave32)
// ---------------------------------------------------------------------------

__device__ __forceinline__ v8f wmma_f16(v16h a, v16h b, v8f c) {
  return __builtin_amdgcn_wmma_f32_16x16x32_f16(false, a, false, b, (short)0, c,
                                                false, false);
}

__device__ __forceinline__ v8f zero8() {
  v8f z = {0.f, 0.f, 0.f, 0.f, 0.f, 0.f, 0.f, 0.f};
  return z;
}

// A-fragment (16x32 f16, M x K).  tile points at element (m0, k0), ld in halves.
// Lane L (0-15): row M=L, K = {0..7, 16..23};  lane L+16: row M=L, K = {8..15, 24..31}.
__device__ __forceinline__ v16h load_a_f16(const _Float16* tile, int ld) {
  const int lane = threadIdx.x & 31;
  const int r = lane & 15;
  const int hi = (lane >> 4) & 1;
  const _Float16* p = tile + r * ld + hi * 8;
  v8h c0 = *(const v8h*)(p);
  v8h c1 = *(const v8h*)(p + 16);
  v16h a;
#pragma unroll
  for (int j = 0; j < 8; ++j) { a[j] = c0[j]; a[j + 8] = c1[j]; }
  return a;
}

// B-fragment (32x16 f16, K x N): lane n: K = {0..7, 16..23}; lane n+16: K = {8..15, 24..31}
// (source row n holds that column).  tile points at (n0 row, k0 col); ld in halves.
__device__ __forceinline__ v16h load_b_f16(const _Float16* tile, int ld) {
  const int lane = threadIdx.x & 31;
  const int n = lane & 15;
  const int kh = (lane >> 4) & 1;
  const _Float16* p = tile + n * ld + kh * 8;
  v8h c0 = *(const v8h*)(p);
  v8h c1 = *(const v8h*)(p + 16);
  v16h b;
#pragma unroll
  for (int j = 0; j < 8; ++j) { b[j] = c0[j]; b[j + 8] = c1[j]; }
  return b;
}

// Cross-lane reductions inside each 16-lane half (masks 1,2,4,8 stay in-half).
__device__ __forceinline__ float rmax16(float x) {
#pragma unroll
  for (int m = 1; m <= 8; m <<= 1) x = fmaxf(x, __shfl_xor(x, m, 32));
  return x;
}
__device__ __forceinline__ float rsum16(float x) {
#pragma unroll
  for (int m = 1; m <= 8; m <<= 1) x += __shfl_xor(x, m, 32);
  return x;
}

// ---------------------------------------------------------------------------
// Async staging of one 64(N) x 32(K) f16 weight panel slice into LDS.
// 256 threads, 16B per lane -> one global_load_async_to_lds_b128 per wave.
// Tracked with ASYNCcnt (CDNA5 ISA).
// ---------------------------------------------------------------------------
__device__ __forceinline__ void stage_b_async(const _Float16* __restrict__ W,
                                              int n0, int k0,
                                              _Float16* buf /*LDS [64][32]*/) {
  const int t = threadIdx.x;       // 0..255
  const int row = t >> 2;          // 0..63
  const int chunk = t & 3;         // 0..3 (8 halves each)
  unsigned gofs = (unsigned)(((n0 + row) * DMODEL + k0 + chunk * 8) * 2);
  unsigned lofs = (unsigned)(unsigned long long)(buf + row * 32 + chunk * 8);
  asm volatile("global_load_async_to_lds_b128 %0, %1, %2"
               :: "v"(lofs), "v"(gofs), "s"(W)
               : "memory");
}

__device__ __forceinline__ void wait_async_le1() {
  asm volatile("s_wait_asynccnt 0x1" ::: "memory");
}
__device__ __forceinline__ void wait_async_0() {
  asm volatile("s_wait_asynccnt 0x0" ::: "memory");
}
__device__ __forceinline__ void wait_lds_0() {
  asm volatile("s_wait_dscnt 0x0" ::: "memory");
}

// ---------------------------------------------------------------------------
// Stage 1: f32 -> f16 conversion (vectorized x4)
// ---------------------------------------------------------------------------
__global__ void cvt_f32_to_f16(const float* __restrict__ src,
                               _Float16* __restrict__ dst, int n) {
  int i = (blockIdx.x * blockDim.x + threadIdx.x) * 4;
  if (i < n) {
    float4 f = *(const float4*)(src + i);
    v4h h;
    h[0] = (_Float16)f.x; h[1] = (_Float16)f.y;
    h[2] = (_Float16)f.z; h[3] = (_Float16)f.w;
    *(v4h*)(dst + i) = h;
  }
}

// ---------------------------------------------------------------------------
// Stage 2: projection GEMM  C[M,N] = A[M,K] * W[N,K]^T  (f16 in, f16 out)
// Block: 256 threads = 8 waves; wave tile = 32(M) x 64(N); block = 256 x 64.
// Steady-state loop: unconditional async stage of the next LDS slice +
// s_wait_asynccnt 0x1 (last k-step peeled), unroll-2 so the A/LDS double
// buffers are renamed instead of copied.
// transpose_out=0: out[m*HD + n] (Q/K);  transpose_out=1: out[n*S2 + m] (v_t).
// ---------------------------------------------------------------------------
__global__ __launch_bounds__(256) void proj_gemm_f16(
    const _Float16* __restrict__ A, const _Float16* __restrict__ W,
    _Float16* __restrict__ out, int transpose_out) {
  __shared__ __align__(16) _Float16 bsm[2][64 * 32];  // 2 x 4KB k-slices

  const int lane = threadIdx.x & 31;
  const int wave = threadIdx.x >> 5;
  const int m0 = blockIdx.x * 256 + wave * 32;
  const int n0 = blockIdx.y * 64;

  v8f acc0[4], acc1[4];
#pragma unroll
  for (int t = 0; t < 4; ++t) { acc0[t] = zero8(); acc1[t] = zero8(); }

  // Prologue: stage k-slice 0 and preload A fragments for k0 = 0.
  stage_b_async(W, n0, 0, bsm[0]);
  v16h a0 = load_a_f16(A + m0 * DMODEL, DMODEL);
  v16h a1 = load_a_f16(A + (m0 + 16) * DMODEL, DMODEL);

  // Steady state: 31 of 32 k-steps, unconditional issue + partial wait.
#pragma unroll 2
  for (int it = 0; it < 31; ++it) {
    const int k0 = it * 32;
    stage_b_async(W, n0, k0 + 32, bsm[(it + 1) & 1]);
    wait_async_le1();          // current slice complete (in-order ASYNCcnt)
    __syncthreads();

    v16h an0 = load_a_f16(A + m0 * DMODEL + k0 + 32, DMODEL);
    v16h an1 = load_a_f16(A + (m0 + 16) * DMODEL + k0 + 32, DMODEL);

    const _Float16* bs = bsm[it & 1];
#pragma unroll
    for (int t = 0; t < 4; ++t) {
      v16h b = load_b_f16(bs + (t * 16) * 32, 32);
      acc0[t] = wmma_f16(a0, b, acc0[t]);
      acc1[t] = wmma_f16(a1, b, acc1[t]);
    }
    __syncthreads();           // protect slice before it is overwritten
    a0 = an0; a1 = an1;
  }

  // Epilogue: final k-step (slice lives in bsm[1]).
  wait_async_0();
  __syncthreads();
#pragma unroll
  for (int t = 0; t < 4; ++t) {
    v16h b = load_b_f16(bsm[1] + (t * 16) * 32, 32);
    acc0[t] = wmma_f16(a0, b, acc0[t]);
    acc1[t] = wmma_f16(a1, b, acc1[t]);
  }

  const int r = lane & 15;
  const int hi = (lane >> 4) & 1;
  if (!transpose_out) {
#pragma unroll
    for (int t = 0; t < 4; ++t)
#pragma unroll
      for (int i = 0; i < 8; ++i) {
        out[(m0 + i + 8 * hi) * HD + n0 + t * 16 + r] = (_Float16)acc0[t][i];
        out[(m0 + 16 + i + 8 * hi) * HD + n0 + t * 16 + r] = (_Float16)acc1[t][i];
      }
  } else {
#pragma unroll
    for (int t = 0; t < 4; ++t) {
      v8h p0, p1;
#pragma unroll
      for (int i = 0; i < 8; ++i) { p0[i] = (_Float16)acc0[t][i]; p1[i] = (_Float16)acc1[t][i]; }
      *(v8h*)(out + (n0 + t * 16 + r) * S2 + m0 + 8 * hi) = p0;       // 16B stores
      *(v8h*)(out + (n0 + t * 16 + r) * S2 + m0 + 16 + 8 * hi) = p1;
    }
  }
}

// ---------------------------------------------------------------------------
// Stage 3: flash attention.  One wave per (head, 16-query block).
// Next key block's K fragments prefetched into registers; per-wave private
// LDS slice for the P transpose (s_wait_dscnt only; no workgroup barrier).
// ---------------------------------------------------------------------------
__global__ __launch_bounds__(128) void flash_attn_f16(
    const _Float16* __restrict__ qh, const _Float16* __restrict__ kh,
    const _Float16* __restrict__ vt, _Float16* __restrict__ oh) {
  __shared__ __align__(16) _Float16 pbuf[4][16 * 32];  // per-wave staging

  const int lane = threadIdx.x & 31;
  const int wave = threadIdx.x >> 5;
  const int task = blockIdx.x * 4 + wave;  // 2048 = 16 heads * 128 q-blocks
  const int h = task >> 7;
  const int q0 = (task & 127) * 16;
  const int r = lane & 15;
  const int hi = (lane >> 4) & 1;

  const v16h aq0 = load_a_f16(qh + q0 * HD + h * DHEAD + 0, HD);
  const v16h aq1 = load_a_f16(qh + q0 * HD + h * DHEAD + 32, HD);

  float m[8], l[8];
#pragma unroll
  for (int i = 0; i < 8; ++i) { m[i] = -1.0e30f; l[i] = 0.f; }
  v8f o0 = zero8(), o1 = zero8(), o2 = zero8(), o3 = zero8();

  _Float16* pb = pbuf[wave];
  const float sc = 0.125f;  // 1/sqrt(DK)

  // Preload K fragments for block 0.
  v16h bk00 = load_b_f16(kh + 0 * HD + h * DHEAD + 0, HD);
  v16h bk01 = load_b_f16(kh + 0 * HD + h * DHEAD + 32, HD);
  v16h bk10 = load_b_f16(kh + 16 * HD + h * DHEAD + 0, HD);
  v16h bk11 = load_b_f16(kh + 16 * HD + h * DHEAD + 32, HD);

#pragma unroll 2
  for (int kb = 0; kb < S2; kb += 32) {
    // V fragments for this block (loads overlap with the S WMMAs below).
    v16h bv0 = load_b_f16(vt + (h * DHEAD + 0) * S2 + kb, S2);
    v16h bv1 = load_b_f16(vt + (h * DHEAD + 16) * S2 + kb, S2);
    v16h bv2 = load_b_f16(vt + (h * DHEAD + 32) * S2 + kb, S2);
    v16h bv3 = load_b_f16(vt + (h * DHEAD + 48) * S2 + kb, S2);

    // ---- S = Q K^T for 32 keys ----
    v8f s0 = zero8(), s1 = zero8();
    s0 = wmma_f16(aq0, bk00, s0);
    s0 = wmma_f16(aq1, bk01, s0);
    s1 = wmma_f16(aq0, bk10, s1);
    s1 = wmma_f16(aq1, bk11, s1);

    // Prefetch next block's K fragments (overlaps softmax VALU below).
    const int knx = (kb + 32 < S2) ? kb + 32 : kb;
    bk00 = load_b_f16(kh + knx * HD + h * DHEAD + 0, HD);
    bk01 = load_b_f16(kh + knx * HD + h * DHEAD + 32, HD);
    bk10 = load_b_f16(kh + (knx + 16) * HD + h * DHEAD + 0, HD);
    bk11 = load_b_f16(kh + (knx + 16) * HD + h * DHEAD + 32, HD);

    // ---- online softmax (rows = VGPR idx + 8*lane-half) ----
    float nm[8], fac[8];
#pragma unroll
    for (int i = 0; i < 8; ++i) {
      s0[i] *= sc; s1[i] *= sc;
      float bm = rmax16(fmaxf(s0[i], s1[i]));
      nm[i] = fmaxf(m[i], bm);
      fac[i] = __expf(m[i] - nm[i]);
      m[i] = nm[i];
    }
#pragma unroll
    for (int i = 0; i < 8; ++i) {
      s0[i] = __expf(s0[i] - nm[i]);
      s1[i] = __expf(s1[i] - nm[i]);
      l[i] = l[i] * fac[i] + rsum16(s0[i] + s1[i]);
      o0[i] *= fac[i]; o1[i] *= fac[i]; o2[i] *= fac[i]; o3[i] *= fac[i];
    }

    // ---- transpose P (f32 D-layout) -> f16 A-fragment layout via LDS ----
#pragma unroll
    for (int i = 0; i < 8; ++i) {
      int row = i + 8 * hi;
      pb[row * 32 + r] = (_Float16)s0[i];
      pb[row * 32 + 16 + r] = (_Float16)s1[i];
    }
    wait_lds_0();                       // per-wave: stores done before reads
    v16h ap = load_a_f16(pb, 32);
    wait_lds_0();                       // reads done before next overwrite

    // ---- O += P * V ----
    o0 = wmma_f16(ap, bv0, o0);
    o1 = wmma_f16(ap, bv1, o1);
    o2 = wmma_f16(ap, bv2, o2);
    o3 = wmma_f16(ap, bv3, o3);
  }

#pragma unroll
  for (int i = 0; i < 8; ++i) {
    float inv = (l[i] > 0.f) ? (1.f / l[i]) : 0.f;
    o0[i] *= inv; o1[i] *= inv; o2[i] *= inv; o3[i] *= inv;
  }
#pragma unroll
  for (int i = 0; i < 8; ++i) {
    int row = q0 + i + 8 * hi;
    oh[row * HD + h * DHEAD + 0 + r]  = (_Float16)o0[i];
    oh[row * HD + h * DHEAD + 16 + r] = (_Float16)o1[i];
    oh[row * HD + h * DHEAD + 32 + r] = (_Float16)o2[i];
    oh[row * HD + h * DHEAD + 48 + r] = (_Float16)o3[i];
  }
}

// ---------------------------------------------------------------------------
// Stage 4: output projection  out = LeakyReLU(attn @ Wo^T + bo)  (f32 out)
// Register double-buffered A and B fragments; last k-step peeled.
// ---------------------------------------------------------------------------
__global__ __launch_bounds__(256) void out_gemm_f16(
    const _Float16* __restrict__ A, const _Float16* __restrict__ Wo,
    const float* __restrict__ bo, float* __restrict__ out) {
  const int lane = threadIdx.x & 31;
  const int wave = threadIdx.x >> 5;
  const int m0 = blockIdx.x * 128 + wave * 16;
  const int n0 = blockIdx.y * 64;

  v8f acc[4];
#pragma unroll
  for (int t = 0; t < 4; ++t) acc[t] = zero8();

  v16h a_c = load_a_f16(A + m0 * HD, HD);
  v16h b_c[4];
#pragma unroll
  for (int t = 0; t < 4; ++t) b_c[t] = load_b_f16(Wo + (n0 + t * 16) * HD, HD);

#pragma unroll 2
  for (int k0 = 0; k0 < HD - 32; k0 += 32) {
    const int kn = k0 + 32;
    v16h a_n = load_a_f16(A + m0 * HD + kn, HD);
    v16h b_n[4];
#pragma unroll
    for (int t = 0; t < 4; ++t) b_n[t] = load_b_f16(Wo + (n0 + t * 16) * HD + kn, HD);
#pragma unroll
    for (int t = 0; t < 4; ++t) acc[t] = wmma_f16(a_c, b_c[t], acc[t]);
    a_c = a_n;
#pragma unroll
    for (int t = 0; t < 4; ++t) b_c[t] = b_n[t];
  }
#pragma unroll
  for (int t = 0; t < 4; ++t) acc[t] = wmma_f16(a_c, b_c[t], acc[t]);

  const int r = lane & 15;
  const int hi = (lane >> 4) & 1;
#pragma unroll
  for (int t = 0; t < 4; ++t) {
    float bias = bo[n0 + t * 16 + r];
#pragma unroll
    for (int i = 0; i < 8; ++i) {
      float v = acc[t][i] + bias;
      v = (v >= 0.f) ? v : 0.01f * v;  // LeakyReLU(0.01)
      out[(m0 + i + 8 * hi) * DMODEL + n0 + t * 16 + r] = v;
    }
  }
}

// ---------------------------------------------------------------------------
// Host-side launcher. Workspace layout (16B-aligned, ~32 MiB total).
// ---------------------------------------------------------------------------
extern "C" void kernel_launch(void* const* d_in, const int* in_sizes, int n_in,
                              void* d_out, int out_size, void* d_ws, size_t ws_size,
                              hipStream_t stream) {
  const float* x1 = (const float*)d_in[0];
  const float* x2 = (const float*)d_in[1];
  const float* Wq = (const float*)d_in[2];
  const float* Wk = (const float*)d_in[3];
  const float* Wv = (const float*)d_in[4];
  const float* Wo = (const float*)d_in[5];
  const float* bo = (const float*)d_in[6];
  float* out = (float*)d_out;

  char* ws = (char*)d_ws;
  _Float16* x1h = (_Float16*)ws;                 ws += (size_t)S1 * DMODEL * 2;
  _Float16* x2h = (_Float16*)ws;                 ws += (size_t)S2 * DMODEL * 2;
  _Float16* Wqh = (_Float16*)ws;                 ws += (size_t)HD * DMODEL * 2;
  _Float16* Wkh = (_Float16*)ws;                 ws += (size_t)HD * DMODEL * 2;
  _Float16* Wvh = (_Float16*)ws;                 ws += (size_t)HD * DMODEL * 2;
  _Float16* Woh = (_Float16*)ws;                 ws += (size_t)DMODEL * HD * 2;
  _Float16* q_h = (_Float16*)ws;                 ws += (size_t)S1 * HD * 2;
  _Float16* k_h = (_Float16*)ws;                 ws += (size_t)S2 * HD * 2;
  _Float16* v_t = (_Float16*)ws;                 ws += (size_t)HD * S2 * 2;
  _Float16* a_h = (_Float16*)ws;                 ws += (size_t)S1 * HD * 2;

  const int NB = 256;
  auto cvt = [&](const float* s, _Float16* d, int n) {
    cvt_f32_to_f16<<<(n / 4 + NB - 1) / NB, NB, 0, stream>>>(s, d, n);
  };
  cvt(x1, x1h, S1 * DMODEL);
  cvt(x2, x2h, S2 * DMODEL);
  cvt(Wq, Wqh, HD * DMODEL);
  cvt(Wk, Wkh, HD * DMODEL);
  cvt(Wv, Wvh, HD * DMODEL);
  cvt(Wo, Woh, DMODEL * HD);

  // Q/K/V projections (V stored transposed for the P*V B-fragments).
  dim3 ggrid(S1 / 256, HD / 64);
  proj_gemm_f16<<<ggrid, 256, 0, stream>>>(x1h, Wqh, q_h, 0);
  proj_gemm_f16<<<ggrid, 256, 0, stream>>>(x2h, Wkh, k_h, 0);
  proj_gemm_f16<<<ggrid, 256, 0, stream>>>(x2h, Wvh, v_t, 1);

  // Flash attention, one wave per (head, 16-query block).
  flash_attn_f16<<<(NHEAD * (S1 / 16)) / 4, 128, 0, stream>>>(q_h, k_h, v_t, a_h);

  // Output projection + bias + LeakyReLU.
  out_gemm_f16<<<dim3(S1 / 128, DMODEL / 64), 256, 0, stream>>>(a_h, Woh, bo, out);
}
